// MSARowAttentionWithPairBias_309237645616
// MI455X (gfx1250) — matrix operand order repaired and checked by reading, not compiled, not run on hardware
//
#include <hip/hip_runtime.h>
#include <hip/hip_bf16.h>

typedef __bf16 bf16;
typedef __attribute__((ext_vector_type(16))) __bf16 v16bf;
typedef __attribute__((ext_vector_type(8)))  float  v8f;

// ---------------------------------------------------------------- utilities
__device__ inline float wave_sum32(float v) {
#pragma unroll
  for (int m = 16; m >= 1; m >>= 1) v += __shfl_xor(v, m, 32);
  return v;
}
__device__ inline float half_sum16(float v) {  // reduce within 16-lane group
#pragma unroll
  for (int m = 8; m >= 1; m >>= 1) v += __shfl_xor(v, m, 32);
  return v;
}
__device__ inline float half_max16(float v) {
#pragma unroll
  for (int m = 8; m >= 1; m >>= 1) v = fmaxf(v, __shfl_xor(v, m, 32));
  return v;
}

__device__ inline v8f wmma_bf16(v16bf a, v16bf b, v8f c) {
  return __builtin_amdgcn_wmma_f32_16x16x32_bf16(false, a, false, b, (short)0,
                                                 c, false, false);
}

// A-fragment (16x32 bf16), row-major MxK source. Per lane this is one
// contiguous 16-element (32 B) run -> compiler merges to 2x b128.
//   m = lane&15;  kb = (lane>>4)*16 + j*2
__device__ inline v16bf load_a_frag(const bf16* __restrict__ A, int lda,
                                    int row0, int col0, int lane) {
  int m = lane & 15, hi = lane >> 4;
  const bf16* p = A + (size_t)(row0 + m) * lda + col0;
  v16bf a;
#pragma unroll
  for (int j = 0; j < 8; ++j) {
    int kb = hi * 16 + (j << 1);
    a[2 * j]     = p[kb];
    a[2 * j + 1] = p[kb + 1];
  }
  return a;
}

// B-fragment (32x16 bf16) from a TRANSPOSED (N x K row-major) source:
// per lane 16 contiguous bf16 -> 2x b128.
//   n = lane&15;  K rows (lane>>4)*16 + (0..15) starting at k0
__device__ inline v16bf load_bt_frag(const bf16* __restrict__ Bt, int ldk,
                                     int k0, int n0, int lane) {
  int n = lane & 15, hi = lane >> 4;
  const bf16* p = Bt + (size_t)(n0 + n) * ldk + k0 + hi * 16;
  v16bf b;
#pragma unroll
  for (int i = 0; i < 16; ++i) b[i] = p[i];
  return b;
}

// ---------------------------------------------------------------- LN kernels
// One wave per 256-channel row: mn = LN(m) -> bf16
__global__ void __launch_bounds__(256)
k_ln_rows(const float* __restrict__ x, const float* __restrict__ w,
          const float* __restrict__ b, bf16* __restrict__ out) {
  int wave = blockIdx.x * (blockDim.x >> 5) + (threadIdx.x >> 5);
  int lane = threadIdx.x & 31;
  const float* row = x + (size_t)wave * 256;
  float v[8];
  float mu = 0.f;
#pragma unroll
  for (int i = 0; i < 8; ++i) { v[i] = row[lane + 32 * i]; mu += v[i]; }
  mu = wave_sum32(mu) * (1.0f / 256.0f);
  float var = 0.f;
#pragma unroll
  for (int i = 0; i < 8; ++i) { float d = v[i] - mu; var += d * d; }
  var = wave_sum32(var) * (1.0f / 256.0f);
  float inv = rsqrtf(var + 1e-5f);
  bf16* orow = out + (size_t)wave * 256;
#pragma unroll
  for (int i = 0; i < 8; ++i) {
    int c = lane + 32 * i;
    orow[c] = (bf16)((v[i] - mu) * inv * w[c] + b[c]);
  }
}

// One wave per pair position (i,j): bias[h][i][j] = LN(z[i,j,:]) . w_z[:,h]
__global__ void __launch_bounds__(256)
k_pair_bias(const float* __restrict__ z, const float* __restrict__ w,
            const float* __restrict__ b, const float* __restrict__ wz,
            float* __restrict__ bias) {
  int pos  = blockIdx.x * (blockDim.x >> 5) + (threadIdx.x >> 5);  // i*256+j
  int lane = threadIdx.x & 31;
  const float* row = z + (size_t)pos * 128;
  float v[4];
  float mu = 0.f;
#pragma unroll
  for (int i = 0; i < 4; ++i) { v[i] = row[lane + 32 * i]; mu += v[i]; }
  mu = wave_sum32(mu) * (1.0f / 128.0f);
  float var = 0.f;
#pragma unroll
  for (int i = 0; i < 4; ++i) { float d = v[i] - mu; var += d * d; }
  var = wave_sum32(var) * (1.0f / 128.0f);
  float inv = rsqrtf(var + 1e-5f);
  float xn[4];
#pragma unroll
  for (int i = 0; i < 4; ++i) {
    int c = lane + 32 * i;
    xn[i] = (v[i] - mu) * inv * w[c] + b[c];
  }
#pragma unroll
  for (int h = 0; h < 8; ++h) {
    float p = 0.f;
#pragma unroll
    for (int i = 0; i < 4; ++i) p += xn[i] * wz[(lane + 32 * i) * 8 + h];
    p = wave_sum32(p);
    if (lane == 0) bias[(size_t)h * 65536 + pos] = p;
  }
}

// Convert a 256x256 f32 weight to TRANSPOSED bf16: out[n*256+k] = in[k*256+n]
__global__ void k_cvt_t(const float* __restrict__ in, bf16* __restrict__ out) {
  int kk = blockIdx.x;    // 0..255 (K row of source)
  int n  = threadIdx.x;   // 0..255 (N col of source)
  out[(size_t)n * 256 + kk] = (bf16)in[(size_t)kk * 256 + n];
}

// ---------------------------------------------------------------- WMMA GEMM
// C[M x 256] = A[M x 256] @ B[256 x 256]; B supplied TRANSPOSED (N x K) bf16.
// Each wave computes a 32x64 tile: 2 A-frags x 4 B-frags -> 8 WMMAs per
// k-step (K=256 -> 8 k-steps -> 64 WMMAs per wave).
// MODE 0: store bf16 raw                    (q, k projections)
// MODE 1: store bf16 sigmoid(x + biasv)     (gate projection)
// MODE 2: store f32  x + biasv  -> d_out    (output projection)
// MODE 3: store bf16 transposed-per-s       (v projection: vT[s][c][r])
template <int MODE>
__global__ void __launch_bounds__(256)
k_gemm_n256(const bf16* __restrict__ A, const bf16* __restrict__ Bt,
            const float* __restrict__ biasv, void* __restrict__ outp) {
  int wave = blockIdx.x * (blockDim.x >> 5) + (threadIdx.x >> 5);
  int lane = threadIdx.x & 31;
  int row0 = (wave >> 2) * 32;   // 2048 row-strips of 32
  int col0 = (wave & 3) * 64;    // 4 col-strips of 64
  v8f acc[2][4] = {};
#pragma unroll
  for (int kk = 0; kk < 8; ++kk) {
    v16bf a0 = load_a_frag(A, 256, row0,      kk * 32, lane);
    v16bf a1 = load_a_frag(A, 256, row0 + 16, kk * 32, lane);
#pragma unroll
    for (int t = 0; t < 4; ++t) {
      v16bf b = load_bt_frag(Bt, 256, kk * 32, col0 + t * 16, lane);
      acc[0][t] = wmma_bf16(a0, b, acc[0][t]);
      acc[1][t] = wmma_bf16(a1, b, acc[1][t]);
    }
  }
  int n = lane & 15, hi = lane >> 4;
#pragma unroll
  for (int i = 0; i < 2; ++i) {
#pragma unroll
    for (int t = 0; t < 4; ++t) {
      int col = col0 + t * 16 + n;
      float bv = biasv ? biasv[col] : 0.0f;
#pragma unroll
      for (int r = 0; r < 8; ++r) {
        int row = row0 + i * 16 + r + 8 * hi;
        float x = acc[i][t][r] + bv;
        if (MODE == 0) {
          ((bf16*)outp)[(size_t)row * 256 + col] = (bf16)x;
        } else if (MODE == 1) {
          ((bf16*)outp)[(size_t)row * 256 + col] =
              (bf16)(1.0f / (1.0f + __expf(-x)));
        } else if (MODE == 2) {
          ((float*)outp)[(size_t)row * 256 + col] = x;
        } else {  // MODE 3: vT[s][c][r], s = row>>8, r = row&255
          ((bf16*)outp)[(((size_t)(row >> 8)) * 256 + col) * 256 + (row & 255)] =
              (bf16)x;
        }
      }
    }
  }
}

// ---------------------------------------------------------------- attention
// grid (S=256, H=8, 4); block 128 = 4 waves; each wave owns 16 q-rows.
// q/k/g layout: [(s*256 + r)*256 + head*32 + c] bf16.
// vT layout:    [(s*256 + c)*256 + r] bf16  (per-s transposed V).
__global__ void __launch_bounds__(128)
k_attn(const bf16* __restrict__ q, const bf16* __restrict__ k,
       const bf16* __restrict__ vT, const bf16* __restrict__ g,
       const float* __restrict__ bias, bf16* __restrict__ go) {
  __shared__ bf16 attn_s[4 * 16 * 256];  // 32 KB: one 16x256 strip per wave

  int s = blockIdx.x, h = blockIdx.y;
  int wave = threadIdx.x >> 5, lane = threadIdx.x & 31;
  int q0 = blockIdx.z * 64 + wave * 16;
  int n = lane & 15, hi = lane >> 4;

  // ---- logits strip: Q(16x32) @ K^T(32x256) -> 16 WMMA tiles
  v16bf aq = load_a_frag(q + ((size_t)s * 256 + q0) * 256, 256, 0, h * 32, lane);

  v8f acc[16];
#pragma unroll
  for (int t = 0; t < 16; ++t) {
    // B[c][nn] = K[t*16+nn][h*32+c]; 16 contiguous bf16 per lane -> 2x b128
    const bf16* kr =
        k + ((size_t)s * 256 + t * 16 + n) * 256 + h * 32 + hi * 16;
    v16bf bkt;
#pragma unroll
    for (int i = 0; i < 16; ++i) bkt[i] = kr[i];
    v8f zero = {};
    acc[t] = wmma_bf16(aq, bkt, zero);
  }

  // ---- bias + scale + softmax over k (row lives in one 16-lane half-group)
  const float scale = 0.17677669529663689f;  // 1/sqrt(32)
  const float* biasH = bias + (size_t)h * 65536;
#pragma unroll
  for (int r = 0; r < 8; ++r) {
    int qrow = q0 + r + 8 * hi;
    const float* bp = biasH + (size_t)qrow * 256 + n;
    float mx = -1e30f;
#pragma unroll
    for (int t = 0; t < 16; ++t) {
      float l = acc[t][r] * scale + bp[t * 16];
      acc[t][r] = l;
      mx = fmaxf(mx, l);
    }
    mx = half_max16(mx);
    float sum = 0.f;
#pragma unroll
    for (int t = 0; t < 16; ++t) {
      float e = __expf(acc[t][r] - mx);
      acc[t][r] = e;
      sum += e;
    }
    sum = half_sum16(sum);
    float inv = 1.0f / sum;
    bf16* arow = attn_s + ((size_t)wave * 16 + r + 8 * hi) * 256;
#pragma unroll
    for (int t = 0; t < 16; ++t) arow[t * 16 + n] = (bf16)(acc[t][r] * inv);
  }

  // ---- O strip: attn(16x256) @ V(256x32), V read via per-s transpose
  const bf16* Vt = vT + (size_t)s * 65536;  // [c][r] for this s
#pragma unroll
  for (int tn2 = 0; tn2 < 2; ++tn2) {
    v8f oacc = {};
#pragma unroll
    for (int kt = 0; kt < 8; ++kt) {
      v16bf aa =
          load_a_frag(attn_s + (size_t)wave * 16 * 256, 256, 0, kt * 32, lane);
      // B[kkLocal][nn] = V[kt*32+kkLocal][h*32+tn2*16+nn] = Vt[c][kRow]:
      // 16 contiguous bf16 per lane -> 2x b128
      const bf16* vp =
          Vt + (size_t)(h * 32 + tn2 * 16 + n) * 256 + kt * 32 + hi * 16;
      v16bf bb;
#pragma unroll
      for (int i = 0; i < 16; ++i) bb[i] = vp[i];
      oacc = wmma_bf16(aa, bb, oacc);
    }
#pragma unroll
    for (int r = 0; r < 8; ++r) {
      int qrow = q0 + r + 8 * hi;
      size_t idx = ((size_t)s * 256 + qrow) * 256 + h * 32 + tn2 * 16 + n;
      go[idx] = (bf16)((float)g[idx] * oacc[r]);
    }
  }
}

// ---------------------------------------------------------------- launcher
extern "C" void kernel_launch(void* const* d_in, const int* in_sizes, int n_in,
                              void* d_out, int out_size, void* d_ws,
                              size_t ws_size, hipStream_t stream) {
  const float* m      = (const float*)d_in[0];
  const float* z      = (const float*)d_in[1];
  const float* ln_m_w = (const float*)d_in[2];
  const float* ln_m_b = (const float*)d_in[3];
  const float* ln_z_w = (const float*)d_in[4];
  const float* ln_z_b = (const float*)d_in[5];
  const float* w_z    = (const float*)d_in[6];
  const float* w_q    = (const float*)d_in[7];
  const float* w_k    = (const float*)d_in[8];
  const float* w_v    = (const float*)d_in[9];
  const float* w_g    = (const float*)d_in[10];
  const float* b_g    = (const float*)d_in[11];
  const float* w_o    = (const float*)d_in[12];
  const float* b_o    = (const float*)d_in[13];

  char* ws = (char*)d_ws;
  size_t off = 0;
  auto alloc = [&](size_t bytes) {
    void* p = ws + off;
    off = (off + bytes + 255) & ~(size_t)255;
    return p;
  };
  const size_t ACT = 256ull * 256ull * 256ull;  // 16.7M elements
  float* bias = (float*)alloc(8ull * 65536ull * sizeof(float));
  bf16* mn  = (bf16*)alloc(ACT * sizeof(bf16));
  bf16* qb  = (bf16*)alloc(ACT * sizeof(bf16));
  bf16* kb  = (bf16*)alloc(ACT * sizeof(bf16));
  bf16* vtb = (bf16*)alloc(ACT * sizeof(bf16));  // per-s transposed V
  bf16* gb  = (bf16*)alloc(ACT * sizeof(bf16));
  bf16* wqt = (bf16*)alloc(65536 * sizeof(bf16));
  bf16* wkt = (bf16*)alloc(65536 * sizeof(bf16));
  bf16* wvt = (bf16*)alloc(65536 * sizeof(bf16));
  bf16* wgt = (bf16*)alloc(65536 * sizeof(bf16));
  bf16* wot = (bf16*)alloc(65536 * sizeof(bf16));
  bf16* go  = mn;  // mn is dead after the 4 projections; reuse for gated O

  // pair bias + LN(m)
  k_pair_bias<<<8192, 256, 0, stream>>>(z, ln_z_w, ln_z_b, w_z, bias);
  k_ln_rows<<<8192, 256, 0, stream>>>(m, ln_m_w, ln_m_b, mn);

  // weights -> transposed bf16 (N x K)
  k_cvt_t<<<256, 256, 0, stream>>>(w_q, wqt);
  k_cvt_t<<<256, 256, 0, stream>>>(w_k, wkt);
  k_cvt_t<<<256, 256, 0, stream>>>(w_v, wvt);
  k_cvt_t<<<256, 256, 0, stream>>>(w_g, wgt);
  k_cvt_t<<<256, 256, 0, stream>>>(w_o, wot);

  // projections (WMMA, 32x64 tiles/wave): q,k raw; v transposed; g sigmoid
  k_gemm_n256<0><<<1024, 256, 0, stream>>>(mn, wqt, nullptr, qb);
  k_gemm_n256<0><<<1024, 256, 0, stream>>>(mn, wkt, nullptr, kb);
  k_gemm_n256<3><<<1024, 256, 0, stream>>>(mn, wvt, nullptr, vtb);
  k_gemm_n256<1><<<1024, 256, 0, stream>>>(mn, wgt, b_g, gb);

  // attention (WMMA logits + WMMA attn@V + gating), writes go (reuses mn)
  k_attn<<<dim3(256, 8, 4), 128, 0, stream>>>(qb, kb, vtb, gb, bias, go);

  // output projection (WMMA) -> f32 d_out
  k_gemm_n256<2><<<1024, 256, 0, stream>>>(go, wot, b_o, (float*)d_out);
}
